// LSTMFeatureExtractor_901943132625
// MI455X (gfx1250) — matrix operand order repaired and checked by reading, not compiled, not run on hardware
//
#include <hip/hip_runtime.h>
#include <hip/hip_bf16.h>
#include <math.h>

// ---------------------------------------------------------------------------
// LSTM feature extractor for MI455X (gfx1250), bf16 WMMA path.
// B=2048, S=256, F=64, H=256, D=128, 4H=1024.
// Grid: B/16 workgroups; each WG owns a 16-row batch tile and scans all S
// steps with (c,h) resident on-chip. Packed bf16 weights stay L2-resident
// and are intentionally re-streamed from L2 every timestep (an opaque asm
// barrier stops LLVM from hoisting the 80 fragments into spilled registers).
// ---------------------------------------------------------------------------

typedef __attribute__((ext_vector_type(16))) __bf16 bf16x16;
typedef __attribute__((ext_vector_type(8)))  __bf16 bf16x8;
typedef __attribute__((ext_vector_type(8)))  float  f32x8;

#define S_LEN 256
#define F_DIM 64
#define H_DIM 256
#define D_DIM 128
#define G_DIM 1024   /* 4*H */
#define BT    16     /* batch tile per workgroup = one WMMA M tile */

#define HSTRIDE 272  /* padded LDS row stride (bf16 elems) for h  */
#define OSTRIDE 80   /* padded LDS row stride (bf16 elems) for x  */

// Load one 16x32 bf16 A-fragment for v_wmma_f32_16x16x32_bf16 from an LDS row.
// Per ISA 7.12.2: lanes 0-15 hold K=[k0..k0+8) and [k0+16..k0+24);
// lanes 16-31 hold K=[k0+8..k0+16) and [k0+24..k0+32). Row M = lane%16.
static __device__ __forceinline__ bf16x16 load_a_frag(const __bf16* rowp, int kbase) {
  union { bf16x16 v; bf16x8 h[2]; } u;
  u.h[0] = *(const bf16x8*)(rowp + kbase);
  u.h[1] = *(const bf16x8*)(rowp + kbase + 16);
  return u.v;
}

// Repack a row-major f32 [K][N] matrix into bf16 WMMA B-fragments.
// Fragment (nt, kt) is 512 bf16, laid out so lane L reads 16 contiguous bf16:
//   k = kt*32 + (L/16)*8 + (e%8) + (e/8)*16,  n = nt*16 + (L%16)
__global__ void pack_b_frags(const float* __restrict__ src,
                             __bf16* __restrict__ dst, int K, int N) {
  int t = blockIdx.x * blockDim.x + threadIdx.x;
  int total = K * N;
  if (t >= total) return;
  int frag   = t >> 9;        // /512
  int within = t & 511;
  int lane   = within >> 4;
  int e      = within & 15;
  int ktcnt  = K >> 5;        // K/32
  int nt     = frag / ktcnt;
  int kt     = frag - nt * ktcnt;
  int k      = kt * 32 + (lane >> 4) * 8 + (e & 7) + (e >> 3) * 16;
  int n      = nt * 16 + (lane & 15);
  dst[t] = (__bf16)src[(size_t)k * N + n];
}

__global__ __launch_bounds__(256)
void lstm_wmma_kernel(const float* __restrict__ obs,
                      const float* __restrict__ bh,
                      const float* __restrict__ bd,
                      const __bf16* __restrict__ WiP,
                      const __bf16* __restrict__ WhP,
                      const __bf16* __restrict__ WdP,
                      float* __restrict__ out) {
  __shared__ __bf16 h_lds[BT * HSTRIDE];
  __shared__ __bf16 x_lds[BT * OSTRIDE];

  const int tid   = threadIdx.x;
  const int w     = tid >> 5;     // wave id 0..7
  const int lane  = tid & 31;
  const int lo    = lane & 15;    // N within tile / row M for A
  const int khalf = lane >> 4;    // K-half select for A, M-half for C
  const int b0    = blockIdx.x * BT;

  // zero initial hidden state
  for (int i = tid; i < BT * HSTRIDE; i += 256) h_lds[i] = (__bf16)0.f;

  // Wave w owns, for every gate q, the two n-tiles 16q + 2w + {0,1}
  // => per-gate columns [32w, 32w+32): matching columns across i/f/g/o,
  // so the cell update is purely register-resident.
  float biasv[4][2];
#pragma unroll
  for (int q = 0; q < 4; ++q)
#pragma unroll
    for (int j = 0; j < 2; ++j)
      biasv[q][j] = bh[q * H_DIM + 32 * w + 16 * j + lo];

  f32x8 cst[2];
#pragma unroll
  for (int j = 0; j < 2; ++j)
#pragma unroll
    for (int e = 0; e < 8; ++e) cst[j][e] = 0.f;

  const float* obs_base = obs + (size_t)b0 * S_LEN * F_DIM;

  // staging coords: 256 threads x 4 contiguous floats = 16x64 tile
  const int srow = tid >> 4;
  const int scol = (tid & 15) * 4;

  for (int s = 0; s < S_LEN; ++s) {
    // ---- stage obs[b-tile, s, :] into LDS as bf16 -----------------------
    {
      const float4* gp = (const float4*)(obs_base +
          (size_t)srow * S_LEN * F_DIM + (size_t)s * F_DIM + scol);
      float4 v = *gp;
      __bf16* dp = x_lds + srow * OSTRIDE + scol;
      dp[0] = (__bf16)v.x; dp[1] = (__bf16)v.y;
      dp[2] = (__bf16)v.z; dp[3] = (__bf16)v.w;
      if (s + 1 < S_LEN)   // prefetch next timestep's tile (global_prefetch_b8)
        __builtin_prefetch((const char*)gp + F_DIM * sizeof(float), 0, 1);
    }
    __syncthreads();

    // Opaque zero offset, rewritten by asm every iteration: stops LLVM from
    // proving the weight-fragment loads loop-invariant and hoisting ~640
    // VGPRs of fragments into scratch spills. Weights are meant to be
    // re-streamed from L2 (they fit easily in 192 MB L2) every step.
    int zofs = 0;
    asm volatile("" : "+s"(zofs));
    const __bf16* whp = WhP + zofs;
    const __bf16* wip = WiP + zofs;

    // Per-(q,j) fragment base: fragments for consecutive kt are contiguous
    // (frag index = nt*KT + kt), so inner loads use constant byte offsets.
    const __bf16* wh_base[4][2];
    const __bf16* wi_base[4][2];
#pragma unroll
    for (int q = 0; q < 4; ++q)
#pragma unroll
      for (int j = 0; j < 2; ++j) {
        int nt = q * 16 + 2 * w + j;
        wh_base[q][j] = whp + (((size_t)nt * (H_DIM / 32)) << 9) + (lane << 4);
        wi_base[q][j] = wip + (((size_t)nt * (F_DIM / 32)) << 9) + (lane << 4);
      }

    // ---- gates = x@Wi + h@Wh + bh (bf16 WMMA, fp32 acc) -----------------
    f32x8 acc[4][2];
#pragma unroll
    for (int q = 0; q < 4; ++q)
#pragma unroll
      for (int j = 0; j < 2; ++j)
#pragma unroll
        for (int e = 0; e < 8; ++e) acc[q][j][e] = biasv[q][j];

    const __bf16* xrow = x_lds + lo * OSTRIDE;
    const __bf16* hrow = h_lds + lo * HSTRIDE;

#pragma unroll
    for (int kt = 0; kt < F_DIM / 32; ++kt) {          // x projection, K=64
      bf16x16 a = load_a_frag(xrow, kt * 32 + khalf * 8);
#pragma unroll
      for (int q = 0; q < 4; ++q)
#pragma unroll
        for (int j = 0; j < 2; ++j) {
          bf16x16 bf = *(const bf16x16*)(wi_base[q][j] + (kt << 9));
          acc[q][j] = __builtin_amdgcn_wmma_f32_16x16x32_bf16(
              false, a, false, bf, (short)0, acc[q][j], false, false);
        }
    }
#pragma unroll
    for (int kt = 0; kt < H_DIM / 32; ++kt) {          // recurrent, K=256
      bf16x16 a = load_a_frag(hrow, kt * 32 + khalf * 8);
#pragma unroll
      for (int q = 0; q < 4; ++q)
#pragma unroll
        for (int j = 0; j < 2; ++j) {
          bf16x16 bf = *(const bf16x16*)(wh_base[q][j] + (kt << 9));
          acc[q][j] = __builtin_amdgcn_wmma_f32_16x16x32_bf16(
              false, a, false, bf, (short)0, acc[q][j], false, false);
        }
    }

    // ---- gate nonlinearities + cell update (register resident) ----------
    f32x8 hv[2];
#pragma unroll
    for (int j = 0; j < 2; ++j) {
#pragma unroll
      for (int e = 0; e < 8; ++e) {
        float iv = 1.f / (1.f + __expf(-acc[0][j][e]));
        float fv = 1.f / (1.f + __expf(-acc[1][j][e]));
        float gv = tanhf(acc[2][j][e]);
        float ov = 1.f / (1.f + __expf(-acc[3][j][e]));
        float cv = fv * cst[j][e] + iv * gv;
        cst[j][e] = cv;
        hv[j][e]  = ov * tanhf(cv);
      }
    }

    // all waves finished reading old h before we overwrite it
    __syncthreads();
#pragma unroll
    for (int j = 0; j < 2; ++j)
#pragma unroll
      for (int v = 0; v < 8; ++v)
        h_lds[(v + 8 * khalf) * HSTRIDE + 32 * w + 16 * j + lo] =
            (__bf16)hv[j][v];
  }
  __syncthreads();

  // ---- head: out = relu(h @ Wd + bd), wave w owns out n-tile w ----------
  {
    f32x8 oacc;
    float bo = bd[w * 16 + lo];
#pragma unroll
    for (int e = 0; e < 8; ++e) oacc[e] = bo;
    const __bf16* hrow = h_lds + lo * HSTRIDE;
#pragma unroll
    for (int kt = 0; kt < H_DIM / 32; ++kt) {
      bf16x16 a  = load_a_frag(hrow, kt * 32 + khalf * 8);
      bf16x16 bf = *(const bf16x16*)(WdP +
          (((size_t)(w * (H_DIM / 32) + kt)) << 9) + (lane << 4));
      oacc = __builtin_amdgcn_wmma_f32_16x16x32_bf16(
          false, a, false, bf, (short)0, oacc, false, false);
    }
#pragma unroll
    for (int v = 0; v < 8; ++v) {
      float val = oacc[v];
      out[(size_t)(b0 + v + 8 * khalf) * D_DIM + w * 16 + lo] =
          val > 0.f ? val : 0.f;
    }
  }
}

extern "C" void kernel_launch(void* const* d_in, const int* in_sizes, int n_in,
                              void* d_out, int out_size, void* d_ws, size_t ws_size,
                              hipStream_t stream) {
  const float* obs = (const float*)d_in[0];   // [B,S,F]
  const float* Wi  = (const float*)d_in[1];   // [F,4H]
  const float* Wh  = (const float*)d_in[2];   // [H,4H]
  const float* bh  = (const float*)d_in[3];   // [4H]
  const float* Wd  = (const float*)d_in[4];   // [H,D]
  const float* bd  = (const float*)d_in[5];   // [D]
  float* out = (float*)d_out;                 // [B,D]

  char* ws = (char*)d_ws;
  __bf16* WhP = (__bf16*)(ws);                       // 512 KB
  __bf16* WiP = (__bf16*)(ws + 512 * 1024);          // 128 KB
  __bf16* WdP = (__bf16*)(ws + 640 * 1024);          //  64 KB

  pack_b_frags<<<(H_DIM * G_DIM + 255) / 256, 256, 0, stream>>>(Wh, WhP, H_DIM, G_DIM);
  pack_b_frags<<<(F_DIM * G_DIM + 255) / 256, 256, 0, stream>>>(Wi, WiP, F_DIM, G_DIM);
  pack_b_frags<<<(H_DIM * D_DIM + 255) / 256, 256, 0, stream>>>(Wd, WdP, H_DIM, D_DIM);

  lstm_wmma_kernel<<<2048 / BT, 256, 0, stream>>>(obs, bh, bd, WiP, WhP, WdP, out);
}
